// DKT_xLSTM_9036611190819
// MI455X (gfx1250) — compile-verified, matrix operand order checked by reading, not run on hardware
//
#include <hip/hip_runtime.h>
#include <math.h>

// ---------------- model dims ----------------
#define E_DIM   100
#define NH      2
#define INNER   200
#define HD      100
#define DH      50
#define KCONV   4
#define FF      192
#define NBLK    7
#define SLSTM_AT 1
#define OUTDIM  100
#define BATCH   1024
#define SEQ     50
#define T_TOK   (BATCH*SEQ)   // 51200 tokens
#define KP_MAX  224           // max padded K (INNER=200 -> 224)

typedef _Float16 half16 __attribute__((ext_vector_type(16)));
typedef float    float8 __attribute__((ext_vector_type(8)));

#define WMMA_F16(acc, a, b) \
  __builtin_amdgcn_wmma_f32_16x16x32_f16(false, (a), false, (b), (short)0, (acc), false, false)

__device__ __forceinline__ float rcpf(float x)  { return __builtin_amdgcn_rcpf(x); }
__device__ __forceinline__ float sigf(float x)  { return rcpf(1.f + __expf(-x)); }
__device__ __forceinline__ float siluf(float x) { return x * sigf(x); }
__device__ __forceinline__ float tanh_fast(float x) {
  // saturates correctly: exp(+inf)->inf, rcp(inf)=0 -> 1 ; exp(-inf)->0 -> -1
  return 1.f - 2.f * rcpf(__expf(2.f * x) + 1.f);
}
__device__ __forceinline__ float logsigf(float x) {
  return (x < 0.f) ? (x - __logf(1.f + __expf(x))) : (-__logf(1.f + __expf(-x)));
}
__device__ __forceinline__ float geluf(float x) {
  const float c = 0.7978845608028654f;
  return 0.5f * x * (1.f + tanh_fast(c * (x + 0.044715f * x * x * x)));
}

// f16 half-index -> K offset within a 32-wide K step (CDNA5 16-bit A/B layout):
// lanes 0-15: halves 0..7 -> K 0..7, halves 8..15 -> K 16..23; lanes 16-31: +8.
__device__ __forceinline__ int wmma_koff(int t, int lane) {
  return ((t < 8) ? t : (t + 8)) + ((lane >> 4) << 3);
}

// ============ WMMA GEMM: out[M,N] = A[M,K] @ W[K,N] (+bias)(+resid)(+act) ============
// Block: 256 thr = 8 waves. Block tile: 16 (M) x 128 (N); A strip staged once in LDS
// (f16, zero K-pad), shared by all 8 waves. B streamed from global with clamped
// addresses (no predication): OOB columns are never stored, K-tail is killed by A==0.
__global__ void __launch_bounds__(256)
gemm_f16wmma(const float* __restrict__ A, const float* __restrict__ W,
             const float* __restrict__ bias, const float* __restrict__ resid,
             float* __restrict__ Cout, int M, int N, int K, int act) {
  __shared__ _Float16 sa[16][KP_MAX];
  const int m0 = blockIdx.y << 4;
  const int tid = threadIdx.x;
  const int Kp = (K + 31) & ~31;

  // cooperative A staging: 16 rows x Kp, zero-padded
  for (int r = 0; r < 16; ++r)
    for (int k = tid; k < Kp; k += 256)
      sa[r][k] = (k < K) ? (_Float16)A[(size_t)(m0 + r) * K + k] : (_Float16)0.f;
  __syncthreads();

  const int lane = tid & 31, wave = tid >> 5;
  const int n0 = (blockIdx.x << 7) + (wave << 4);
  if (n0 >= N) return;                    // whole wave exits; no barriers below
  const int bn  = n0 + (lane & 15);
  const int bnc = (bn < N) ? bn : (N - 1);     // clamped column address
  const int arow = lane & 15;
  const int khi  = (lane >> 4) << 3;

  float8 acc = {};
  for (int kb = 0; kb < Kp; kb += 32) {
    half16 a, b;
#pragma unroll
    for (int t = 0; t < 16; ++t) {
      const int k  = kb + ((t < 8) ? t : (t + 8)) + khi;
      const int kc = (k < K) ? k : (K - 1);      // clamp only; no masking needed
      a[t] = sa[arow][k];
      b[t] = (_Float16)W[(size_t)kc * N + bnc];
    }
    acc = WMMA_F16(acc, a, b);
  }

  if (bn < N) {
    const float bv = bias ? bias[bn] : 0.f;      // per-column, hoisted
    float rv[8];
    if (resid) {
#pragma unroll
      for (int j = 0; j < 8; ++j)               // batched loads, one wait
        rv[j] = resid[(size_t)(m0 + j + ((lane >> 4) << 3)) * N + bn];
    } else {
#pragma unroll
      for (int j = 0; j < 8; ++j) rv[j] = 0.f;
    }
#pragma unroll
    for (int j = 0; j < 8; ++j) {
      const int m = m0 + j + ((lane >> 4) << 3);
      float v = acc[j] + bv + rv[j];
      if (act == 1) v = sigf(v);
      Cout[(size_t)m * N + bn] = v;
    }
  }
}

// ============ depthwise causal conv1d + SiLU ============
__global__ void conv_silu(const float* __restrict__ in, int irs, int C,
                          const float* __restrict__ w, const float* __restrict__ b,
                          float* __restrict__ out) {
  size_t idx = (size_t)blockIdx.x * blockDim.x + threadIdx.x;
  if (idx >= (size_t)T_TOK * C) return;
  const int c = (int)(idx % C);
  const size_t tok = idx / C;
  const int s = (int)(tok % SEQ);
  const size_t bs0 = tok - s;
  float acc = b[c];
#pragma unroll
  for (int k = 0; k < KCONV; ++k) {
    const int sp = s - (KCONV - 1) + k;
    if (sp >= 0) acc += in[(bs0 + sp) * irs + c] * w[c * KCONV + k];
  }
  out[idx] = siluf(acc);
}

// ============ block-diagonal q/k/v projection (50 blocks of 4) ============
__global__ void qkv_proj(const float* __restrict__ xc, const float* __restrict__ up,
                         const float* __restrict__ Wq, const float* __restrict__ Wk,
                         const float* __restrict__ Wv,
                         float* __restrict__ q, float* __restrict__ k, float* __restrict__ v) {
  size_t idx = (size_t)blockIdx.x * blockDim.x + threadIdx.x;
  if (idx >= (size_t)T_TOK * INNER) return;
  const int c = (int)(idx % INNER);
  const size_t tok = idx / INNER;
  const int nb = c >> 2, o = c & 3;
  const float* xr = xc + tok * INNER + (nb << 2);        // conv path (q,k)
  const float* vr = up + tok * (2 * INNER) + (nb << 2);  // raw x_in (v)
  float aq = 0.f, ak = 0.f, av = 0.f;
#pragma unroll
  for (int i = 0; i < 4; ++i) {
    const int wi = (nb * 4 + o) * 4 + i;
    aq += xr[i] * Wq[wi];
    ak += xr[i] * Wk[wi];
    av += vr[i] * Wv[wi];
  }
  q[idx] = aq; k[idx] = ak; v[idx] = av;
}

// ============ mLSTM input/forget gates: [q,k,v](600) @ W(600,2) + b ============
__global__ void mgates(const float* __restrict__ q, const float* __restrict__ k,
                       const float* __restrict__ v,
                       const float* __restrict__ Wig, const float* __restrict__ big,
                       const float* __restrict__ Wfg, const float* __restrict__ bfg,
                       float* __restrict__ gsm) {
  size_t idx = (size_t)blockIdx.x * blockDim.x + threadIdx.x;
  if (idx >= (size_t)T_TOK * 4) return;
  const int e = (int)(idx & 3);          // 0,1 = ig(h); 2,3 = fg(h)
  const size_t tok = idx >> 2;
  const int h = e & 1;
  const float* W = (e < 2) ? Wig : Wfg;
  float acc = (e < 2) ? big[h] : bfg[h];
  const float* qr = q + tok * INNER;
  const float* kr = k + tok * INNER;
  const float* vr = v + tok * INNER;
  for (int j = 0; j < INNER; ++j) acc += qr[j] * W[j * NH + h];
  for (int j = 0; j < INNER; ++j) acc += kr[j] * W[(INNER + j) * NH + h];
  for (int j = 0; j < INNER; ++j) acc += vr[j] * W[(2 * INNER + j) * NH + h];
  gsm[tok * 4 + e] = acc;
}

// ============ mLSTM attention core, one workgroup per (b,h), WMMA from LDS ============
__global__ void __launch_bounds__(256)
mlstm_attn(const float* __restrict__ q, const float* __restrict__ k,
           const float* __restrict__ v, const float* __restrict__ gsm,
           float* __restrict__ att) {
  __shared__ _Float16 sq[64][128];   // q rows (s), K-padded to 128, zero pad
  __shared__ _Float16 sk[64][128];   // k rows (t)
  __shared__ _Float16 sv[64][112];   // v rows (t), N-padded to 112
  __shared__ float    scs[64][64];   // Cs (then normalized)
  __shared__ float    fcum[SEQ], iga[SEQ], mrow[SEQ], em[SEQ];

  const int bh = blockIdx.x;
  const int b = bh >> 1, h = bh & 1;
  const int tid = threadIdx.x;
  const size_t base = (size_t)b * SEQ;

  for (int idx = tid; idx < 64 * 128; idx += 256) {
    const int s = idx >> 7, d = idx & 127;
    _Float16 qa = (_Float16)0.f, ka = (_Float16)0.f;
    if (s < SEQ && d < HD) {
      qa = (_Float16)q[(base + s) * INNER + h * HD + d];
      ka = (_Float16)k[(base + s) * INNER + h * HD + d];
    }
    sq[s][d] = qa; sk[s][d] = ka;
  }
  for (int idx = tid; idx < 64 * 112; idx += 256) {
    const int s = idx / 112, d = idx % 112;
    sv[s][d] = (s < SEQ && d < HD) ? (_Float16)v[(base + s) * INNER + h * HD + d]
                                   : (_Float16)0.f;
  }
  if (tid == 0) {  // serial gate prep, S=50: fcum, m[s]=fcum[s]+cummax(ig-fcum)
    float fc = 0.f, cmax = -3.4e38f;
    for (int s = 0; s < SEQ; ++s) {
      const float ig = gsm[(base + s) * 4 + h];
      const float fg = gsm[(base + s) * 4 + 2 + h];
      fc += logsigf(fg);
      fcum[s] = fc; iga[s] = ig;
      cmax = fmaxf(cmax, ig - fc);
      const float m = fc + cmax;
      mrow[s] = m; em[s] = __expf(-m);
    }
  }
  __syncthreads();

  const int lane = tid & 31, wave = tid >> 5;
  const float scale = 0.1f;  // HD^-0.5

  // Cs = (q @ k^T) * scale * Dn   (4x4 tiles of 16, K = 128 padded)
  for (int tile = wave; tile < 16; tile += 8) {
    const int ts = tile >> 2, tt = tile & 3;
    float8 acc = {};
    for (int kb = 0; kb < 128; kb += 32) {
      half16 a, bb;
#pragma unroll
      for (int t = 0; t < 16; ++t) {
        const int kk = kb + wmma_koff(t, lane);
        a[t]  = sq[ts * 16 + (lane & 15)][kk];
        bb[t] = sk[tt * 16 + (lane & 15)][kk];
      }
      acc = WMMA_F16(acc, a, bb);
    }
#pragma unroll
    for (int j = 0; j < 8; ++j) {
      const int s = ts * 16 + j + ((lane >> 4) << 3);
      const int t = tt * 16 + (lane & 15);
      float val = 0.f;
      if (s < SEQ && t <= s)
        val = acc[j] * scale * __expf(fcum[s] - fcum[t] + iga[t] - mrow[s]);
      scs[s][t] = val;
    }
  }
  __syncthreads();

  // n[s] = max(|sum_t Cs|, exp(-m));  normalize rows in LDS
  for (int s = tid; s < SEQ; s += 256) {
    float sum = 0.f;
    for (int t = 0; t < 64; ++t) sum += scs[s][t];
    const float inv = rcpf(fmaxf(fabsf(sum), em[s]));
    for (int t = 0; t < 64; ++t) scs[s][t] *= inv;
  }
  __syncthreads();

  // h = Csn @ v   (4 x 7 tiles, K = 64)
  for (int tile = wave; tile < 28; tile += 8) {
    const int ts = tile / 7, td = tile % 7;
    float8 acc = {};
    for (int kb = 0; kb < 64; kb += 32) {
      half16 a, bb;
#pragma unroll
      for (int t = 0; t < 16; ++t) {
        const int kk = kb + wmma_koff(t, lane);
        a[t]  = (_Float16)scs[ts * 16 + (lane & 15)][kk];
        bb[t] = sv[kk][td * 16 + (lane & 15)];
      }
      acc = WMMA_F16(acc, a, bb);
    }
#pragma unroll
    for (int j = 0; j < 8; ++j) {
      const int s = ts * 16 + j + ((lane >> 4) << 3);
      const int d = td * 16 + (lane & 15);
      if (s < SEQ && d < HD) att[(base + s) * INNER + h * HD + d] = acc[j];
    }
  }
}

// ============ group/layer norm (weight-only affine), one thread per (token,group) ============
__global__ void groupnorm(const float* __restrict__ in, const float* __restrict__ w,
                          float* __restrict__ out, int C, int G) {
  const int D = C / G;
  size_t idx = (size_t)blockIdx.x * blockDim.x + threadIdx.x;
  if (idx >= (size_t)T_TOK * G) return;
  const int g = (int)(idx % G);
  const size_t tok = idx / G;
  const float invD = rcpf((float)D);
  const float* xr = in + tok * C + g * D;
  float m = 0.f;
  for (int d = 0; d < D; ++d) m += xr[d];
  m *= invD;
  float vv = 0.f;
  for (int d = 0; d < D; ++d) { const float t = xr[d] - m; vv += t * t; }
  vv *= invD;
  const float inv = rsqrtf(vv + 1e-5f);
  float* orow = out + tok * C + g * D;
  for (int d = 0; d < D; ++d) orow[d] = (xr[d] - m) * inv * w[g * D + d];
}

// ============ mLSTM combine: (normed + skip*xc) * silu(z) ============
__global__ void mlstm_combine(const float* __restrict__ hn, const float* __restrict__ xc,
                              const float* __restrict__ up, const float* __restrict__ skip,
                              float* __restrict__ out) {
  size_t idx = (size_t)blockIdx.x * blockDim.x + threadIdx.x;
  if (idx >= (size_t)T_TOK * INNER) return;
  const int c = (int)(idx % INNER);
  const size_t tok = idx / INNER;
  const float z = up[tok * (2 * INNER) + INNER + c];
  out[idx] = (hn[idx] + skip[c] * xc[idx]) * siluf(z);
}

// ============ sLSTM headwise gate projections ============
__global__ void sgates(const float* __restrict__ yc, const float* __restrict__ y,
                       const float* __restrict__ Wg, float* __restrict__ g4) {
  size_t idx = (size_t)blockIdx.x * blockDim.x + threadIdx.x;
  if (idx >= (size_t)4 * T_TOK * E_DIM) return;
  const int he = (int)(idx % E_DIM);
  const size_t r = idx / E_DIM;
  const size_t tok = r % T_TOK;
  const int g = (int)(r / T_TOK);
  const int h = he / DH, o = he % DH;
  const float* src = ((g < 2) ? yc : y) + tok * E_DIM + h * DH;
  const float* wr = Wg + ((size_t)((g * NH + h) * DH + o)) * DH;
  float acc = 0.f;
  for (int d = 0; d < DH; ++d) acc += src[d] * wr[d];
  g4[idx] = acc;
}

// ============ sLSTM recurrence: one workgroup per batch element ============
__global__ void __launch_bounds__(128)
slstm_rec(const float* __restrict__ g4, const float* __restrict__ R,
          const float* __restrict__ bias, float* __restrict__ hs) {
  __shared__ float hst[E_DIM], cst[E_DIM], nst[E_DIM], mst[E_DIM], rec[4 * E_DIM];
  const int b = blockIdx.x, tid = threadIdx.x;
  const size_t TE = (size_t)T_TOK * E_DIM;
  for (int i = tid; i < E_DIM; i += 128) { hst[i] = 0.f; cst[i] = 0.f; nst[i] = 0.f; mst[i] = 0.f; }
  __syncthreads();
  for (int s = 0; s < SEQ; ++s) {
    for (int i = tid; i < 4 * E_DIM; i += 128) {
      const int g = i / E_DIM, he = i % E_DIM, h = he / DH;
      const float* rr = R + ((size_t)((g * NH + h) * DH + (he % DH))) * DH;
      const float* hh = hst + h * DH;
      float a = 0.f;
      for (int d = 0; d < DH; ++d) a += hh[d] * rr[d];
      rec[i] = a;
    }
    __syncthreads();
    const size_t tok = (size_t)b * SEQ + s;
    for (int i = tid; i < E_DIM; i += 128) {
      const int h = i / DH, d = i % DH;
      const float ir  = g4[0 * TE + tok * E_DIM + i] + rec[0 * E_DIM + i] + bias[(0 * NH + h) * DH + d];
      const float fr  = g4[1 * TE + tok * E_DIM + i] + rec[1 * E_DIM + i] + bias[(1 * NH + h) * DH + d];
      const float zr  = g4[2 * TE + tok * E_DIM + i] + rec[2 * E_DIM + i] + bias[(2 * NH + h) * DH + d];
      const float orr = g4[3 * TE + tok * E_DIM + i] + rec[3 * E_DIM + i] + bias[(3 * NH + h) * DH + d];
      const float lf = logsigf(fr) + mst[i];
      const float mn = fmaxf(ir, lf);
      const float ii = __expf(ir - mn), ff = __expf(lf - mn);
      const float cn = ff * cst[i] + ii * tanh_fast(zr);
      const float nn = ff * nst[i] + ii;
      const float hn = sigf(orr) * cn * rcpf(nn);
      cst[i] = cn; nst[i] = nn; mst[i] = mn; hst[i] = hn;
      hs[tok * E_DIM + i] = hn;
    }
    __syncthreads();
  }
}

// ============ FFN activation: gelu(gate) * u ============
__global__ void ffn_act(const float* __restrict__ up, float* __restrict__ out) {
  size_t idx = (size_t)blockIdx.x * blockDim.x + threadIdx.x;
  if (idx >= (size_t)T_TOK * FF) return;
  const int j = (int)(idx % FF);
  const size_t tok = idx / FF;
  out[idx] = geluf(up[tok * (2 * FF) + j]) * up[tok * (2 * FF) + FF + j];
}

__global__ void addk(float* __restrict__ dst, const float* __restrict__ src, size_t n) {
  size_t idx = (size_t)blockIdx.x * blockDim.x + threadIdx.x;
  if (idx < n) dst[idx] += src[idx];
}

// ==================== host launcher ====================
static inline unsigned gsz(size_t n, int bs) { return (unsigned)((n + bs - 1) / bs); }

extern "C" void kernel_launch(void* const* d_in, const int* in_sizes, int n_in,
                              void* d_out, int out_size, void* d_ws, size_t ws_size,
                              hipStream_t stream) {
  (void)in_sizes; (void)out_size; (void)ws_size;
  if (n_in < 112) return;  // expect 112 pytree leaves (sorted dict keys)
  auto F = [&](int i) { return (const float*)d_in[i]; };
  // leaf order: Wfc, bfc, blocks[0..6]{sorted}, post_norm_w, x
  const float* Wfc = F(0);
  const float* bfc = F(1);
  const float* post_w = F(110);
  const float* x = F(111);

  const size_t T = T_TOK;
  float* ws  = (float*)d_ws;
  float* h   = ws;                 // T*E   residual stream
  float* y   = h  + T * E_DIM;     // T*E   normed input
  float* up  = y  + T * E_DIM;     // T*400
  float* xc  = up + T * 400;       // T*200
  float* q   = xc + T * 200;       // T*200
  float* kk  = q  + T * 200;       // T*200
  float* v   = kk + T * 200;       // T*200
  float* gsm = v  + T * 200;       // T*4
  float* att = gsm + T * 4;        // T*200
  float* g4  = up;                 // alias: 4*T*E == T*400
  float* yc  = xc;                 // alias
  float* hs  = att;                // alias
  float* ffa = q;                  // alias: T*FF <= T*200

  hipMemcpyAsync(h, x, T * E_DIM * sizeof(float), hipMemcpyDeviceToDevice, stream);

  auto gemm = [&](const float* A, const float* W, const float* bias, const float* resid,
                  float* out, int M, int N, int K, int act) {
    dim3 grid((unsigned)((N + 127) / 128), (unsigned)(M / 16));
    gemm_f16wmma<<<grid, 256, 0, stream>>>(A, W, bias, resid, out, M, N, K, act);
  };

  for (int i = 0; i < NBLK; ++i) {
    if (i == SLSTM_AT) {
      const int bi = 2 + SLSTM_AT * 16;  // = 18
      const float* R      = F(bi + 0);
      const float* Wffd   = F(bi + 1);
      const float* Wffu   = F(bi + 2);
      const float* Wg     = F(bi + 3);
      const float* bgate  = F(bi + 4);
      const float* bffd   = F(bi + 5);
      const float* bffu   = F(bi + 6);
      const float* conv_b = F(bi + 7);
      const float* conv_w = F(bi + 8);
      const float* gn_w   = F(bi + 9);
      const float* ln1    = F(bi + 10);
      const float* ln2    = F(bi + 11);

      groupnorm<<<gsz(T, 256), 256, 0, stream>>>(h, ln1, y, E_DIM, 1);
      conv_silu<<<gsz(T * E_DIM, 256), 256, 0, stream>>>(y, E_DIM, E_DIM, conv_w, conv_b, yc);
      sgates<<<gsz((size_t)4 * T * E_DIM, 256), 256, 0, stream>>>(yc, y, Wg, g4);
      slstm_rec<<<BATCH, 128, 0, stream>>>(g4, R, bgate, hs);
      groupnorm<<<gsz(T * NH, 256), 256, 0, stream>>>(hs, gn_w, hs, E_DIM, NH);
      addk<<<gsz(T * E_DIM, 256), 256, 0, stream>>>(h, hs, T * E_DIM);
      // FFN
      groupnorm<<<gsz(T, 256), 256, 0, stream>>>(h, ln2, y, E_DIM, 1);
      gemm(y, Wffu, bffu, nullptr, up, T_TOK, 2 * FF, E_DIM, 0);
      ffn_act<<<gsz(T * FF, 256), 256, 0, stream>>>(up, ffa);
      gemm(ffa, Wffd, bffd, h, h, T_TOK, E_DIM, FF, 0);
    } else {
      const int bi = (i < SLSTM_AT) ? (2 + i * 16) : (30 + (i - 2) * 16);
      const float* Wdown  = F(bi + 0);
      const float* Wfg    = F(bi + 1);
      const float* Wig    = F(bi + 2);
      const float* Wk     = F(bi + 3);
      const float* Wq     = F(bi + 4);
      const float* Wup    = F(bi + 5);
      const float* Wv     = F(bi + 6);
      const float* bdown  = F(bi + 7);
      const float* bfg    = F(bi + 8);
      const float* big    = F(bi + 9);
      const float* bup    = F(bi + 10);
      const float* conv_b = F(bi + 11);
      const float* conv_w = F(bi + 12);
      const float* ln1    = F(bi + 13);
      const float* mhn_w  = F(bi + 14);
      const float* skip   = F(bi + 15);

      groupnorm<<<gsz(T, 256), 256, 0, stream>>>(h, ln1, y, E_DIM, 1);
      gemm(y, Wup, bup, nullptr, up, T_TOK, 2 * INNER, E_DIM, 0);
      conv_silu<<<gsz(T * INNER, 256), 256, 0, stream>>>(up, 2 * INNER, INNER, conv_w, conv_b, xc);
      qkv_proj<<<gsz(T * INNER, 256), 256, 0, stream>>>(xc, up, Wq, Wk, Wv, q, kk, v);
      mgates<<<gsz(T * 4, 256), 256, 0, stream>>>(q, kk, v, Wig, big, Wfg, bfg, gsm);
      mlstm_attn<<<BATCH * NH, 256, 0, stream>>>(q, kk, v, gsm, att);
      groupnorm<<<gsz(T * NH, 256), 256, 0, stream>>>(att, mhn_w, att, INNER, NH);
      mlstm_combine<<<gsz(T * INNER, 256), 256, 0, stream>>>(att, xc, up, skip, att);
      gemm(att, Wdown, bdown, h, h, T_TOK, E_DIM, INNER, 0);
    }
  }

  groupnorm<<<gsz(T, 256), 256, 0, stream>>>(h, post_w, y, E_DIM, 1);
  gemm(y, Wfc, bfc, nullptr, (float*)d_out, T_TOK, OUTDIM, E_DIM, /*sigmoid*/1);
}
